// GLAWrapper_9285719294259
// MI455X (gfx1250) — matrix operand order repaired and checked by reading, not compiled, not run on hardware
//
#include <hip/hip_runtime.h>

// ---------------------------------------------------------------------------
// Problem constants (from reference)
// ---------------------------------------------------------------------------
#define BB 2
#define TT 4096
#define DD 1024
#define HH 4
#define DK 128
#define DV 256
#define KD 512      // H*DK
#define VD 1024     // H*DV
#define LR 16
#define CC 64       // chunk size
#define NN 64       // T / C
#define ROWS (BB * TT)          // 8192
#define QSCALE 0.08838834764831845f   // DK^-0.5
#define INV_GN (1.0f / 16.0f)

typedef __bf16 bf16;
typedef bf16  bf16x8   __attribute__((ext_vector_type(8)));
typedef bf16  bf16x16  __attribute__((ext_vector_type(16)));
typedef float floatx8  __attribute__((ext_vector_type(8)));
typedef unsigned int u32x4 __attribute__((ext_vector_type(4)));
typedef int i32x4 __attribute__((ext_vector_type(4)));
typedef int i32x8 __attribute__((ext_vector_type(8)));

// ---------------------------------------------------------------------------
// TDM availability (guarded: fallback path is plain cooperative LDS copies)
// ---------------------------------------------------------------------------
#if defined(__has_builtin)
#if __has_builtin(__builtin_amdgcn_tensor_load_to_lds) && \
    __has_builtin(__builtin_amdgcn_s_wait_tensorcnt)
#define HAVE_TDM 1
#endif
#endif
#ifndef HAVE_TDM
#define HAVE_TDM 0
#endif

#if HAVE_TDM
// Build a D# for a 2D f32 tile load (tile_h rows x tile_w elems, row stride in
// elems) and issue TENSOR_LOAD_TO_LDS. Bit packing per CDNA5 ISA:
//  group0: [1:0]=count=1 | lds_addr | global_addr[56:0] | type=2 in [127:126]
//  group1: data_size=2(4B) @[17:16]; tensor_dim0 @[79:48]; tensor_dim1 @[111:80];
//          tile_dim0 @[127:112]; tile_dim1 @[143:128]; dim0_stride @[207:160]
__device__ __forceinline__ void tdm_load_2d_f32(unsigned lds_byte_off,
                                                const float* gptr,
                                                unsigned tile_w, unsigned tile_h,
                                                unsigned row_stride_elems) {
  unsigned long long ga = (unsigned long long)(const void*)gptr;
  u32x4 g0;
  g0.x = 1u;                                 // count=1, user mode
  g0.y = lds_byte_off;                       // D#.lds_addr (bytes)
  g0.z = (unsigned)ga;                       // global_addr[31:0]
  g0.w = ((unsigned)(ga >> 32) & 0x01ffffffu) | (2u << 30);  // addr hi | type=2
  const unsigned d0 = row_stride_elems;      // tensor_dim0 (elems)
  const unsigned d1 = 1u << 20;              // tensor_dim1 (large; no OOB clip)
  const unsigned s0 = row_stride_elems;      // tensor_dim0_stride (elems)
  i32x8 g1;
  g1[0] = (int)(2u << 16);                                   // data_size=4B
  g1[1] = (int)((d0 & 0xffffu) << 16);                       // dim0 lo
  g1[2] = (int)((d0 >> 16) | ((d1 & 0xffffu) << 16));        // dim0 hi|dim1 lo
  g1[3] = (int)((d1 >> 16) | ((tile_w & 0xffffu) << 16));    // dim1 hi|tile_d0
  g1[4] = (int)(tile_h & 0xffffu);                           // tile_d1 (d2=0)
  g1[5] = (int)s0;                                           // dim0_stride lo
  g1[6] = 0;
  g1[7] = 0;
  i32x4 zz = {0, 0, 0, 0};
#if __clang_major__ >= 23
  i32x8 z8 = {0, 0, 0, 0, 0, 0, 0, 0};
  __builtin_amdgcn_tensor_load_to_lds(g0, g1, zz, zz, z8, 0);
#else
  __builtin_amdgcn_tensor_load_to_lds(g0, g1, zz, zz, 0);
#endif
}
#endif  // HAVE_TDM

// ---------------------------------------------------------------------------
// WMMA fragment helpers (CDNA5 16x16x32 bf16, wave32 layouts per ISA 7.12.2)
// ---------------------------------------------------------------------------
__device__ __forceinline__ bf16x16 load_a_frag(const bf16* a, int lda, int m0, int k0, int lane) {
  int row = lane & 15, kh = lane >> 4;
  const bf16* p = a + (m0 + row) * lda + k0 + kh * 8;
  bf16x8 lo = *(const bf16x8*)p;
  bf16x8 hi = *(const bf16x8*)(p + 16);
  return __builtin_shufflevector(lo, hi, 0, 1, 2, 3, 4, 5, 6, 7,
                                         8, 9, 10, 11, 12, 13, 14, 15);
}

__device__ __forceinline__ bf16x16 load_b_frag(const bf16* bt, int ldb, int n0, int k0, int lane) {
  int n = lane & 15, kh = lane >> 4;
  const bf16* p = bt + (n0 + n) * ldb + k0 + kh * 8;
  bf16x8 lo = *(const bf16x8*)p;
  bf16x8 hi = *(const bf16x8*)(p + 16);
  return __builtin_shufflevector(lo, hi, 0, 1, 2, 3, 4, 5, 6, 7,
                                         8, 9, 10, 11, 12, 13, 14, 15);
}

__device__ __forceinline__ floatx8 wmma_bf16(bf16x16 a, bf16x16 b, floatx8 c) {
  return __builtin_amdgcn_wmma_f32_16x16x32_bf16(false, a, false, b, (short)0, c, false, false);
}

// ---------------------------------------------------------------------------
// Cast fp32 -> bf16 (grid-stride)
// ---------------------------------------------------------------------------
__global__ __launch_bounds__(256) void cast_f32_bf16(const float* __restrict__ s,
                                                     bf16* __restrict__ d, long n) {
  long i = (long)blockIdx.x * 256 + threadIdx.x;
  long stride = (long)gridDim.x * 256;
  for (; i < n; i += stride) d[i] = (bf16)s[i];
}

// ---------------------------------------------------------------------------
// Tiled bf16 WMMA GEMM: C[M,N] = A[M,K] * B[K,N], fp32 out.
// Block tile 128(M) x 128(N), 256 threads = 8 waves (2 M-rows x 4 N-cols),
// wave tile 64x32 = 4x2 WMMA tiles (8 wmma / K-step, B-frags reused 4x).
// Double-buffered LDS: one barrier per K-step; global loads for kt+32 overlap
// WMMA on kt. Assumes M%128==0, N%128==0, K%32==0 (true for all calls here).
// ---------------------------------------------------------------------------
__global__ __launch_bounds__(256) void gemm_bf16_wmma(const bf16* __restrict__ A,
                                                      const bf16* __restrict__ B,
                                                      float* __restrict__ C,
                                                      int M, int N, int K) {
  __shared__ bf16 As[2][128 * 32];   // row-major M x K-tile
  __shared__ bf16 Bts[2][128 * 32];  // N-major  N x K-tile (transposed stage-in)

  const int tid  = threadIdx.x;
  const int lane = tid & 31;
  const int wave = tid >> 5;
  const int wm   = wave >> 2;       // 0..1 -> M offset wm*64
  const int wn   = wave & 3;        // 0..3 -> N offset wn*32
  const int m_blk = blockIdx.y * 128;
  const int n_blk = blockIdx.x * 128;

  // stage-in mapping: A 128x32 as 2 chunks/thread of 8 bf16;
  //                   B 32x128 as 2 chunks/thread of 8 contiguous n.
  const int a_row = tid >> 2, a_ko = (tid & 3) * 8;
  const int b_k = tid >> 4, b_n0 = (tid & 15) * 8;
  bf16x8 aReg[2], bReg[2];

  auto g_load = [&](int kt) {
    aReg[0] = *(const bf16x8*)(A + (long)(m_blk + a_row) * K + kt + a_ko);
    aReg[1] = *(const bf16x8*)(A + (long)(m_blk + a_row + 64) * K + kt + a_ko);
    bReg[0] = *(const bf16x8*)(B + (long)(kt + b_k) * N + n_blk + b_n0);
    bReg[1] = *(const bf16x8*)(B + (long)(kt + b_k + 16) * N + n_blk + b_n0);
  };
  auto s_store = [&](int buf) {
    *(bf16x8*)(&As[buf][a_row * 32 + a_ko]) = aReg[0];
    *(bf16x8*)(&As[buf][(a_row + 64) * 32 + a_ko]) = aReg[1];
#pragma unroll
    for (int i = 0; i < 8; ++i) {
      Bts[buf][(b_n0 + i) * 32 + b_k] = bReg[0][i];
      Bts[buf][(b_n0 + i) * 32 + b_k + 16] = bReg[1][i];
    }
  };

  floatx8 acc[4][2] = {};
  g_load(0);
  s_store(0);
  int buf = 0;
  for (int kt = 0; kt < K; kt += 32) {
    __syncthreads();
    const bool more = (kt + 32 < K);
    if (more) g_load(kt + 32);         // overlap next global fetch with WMMA
    bf16x16 bfr0 = load_b_frag(Bts[buf], 32, wn * 32, 0, lane);
    bf16x16 bfr1 = load_b_frag(Bts[buf], 32, wn * 32 + 16, 0, lane);
#pragma unroll
    for (int i = 0; i < 4; ++i) {
      bf16x16 afr = load_a_frag(As[buf], 32, wm * 64 + i * 16, 0, lane);
      acc[i][0] = wmma_bf16(afr, bfr0, acc[i][0]);
      acc[i][1] = wmma_bf16(afr, bfr1, acc[i][1]);
    }
    if (more) s_store(buf ^ 1);
    buf ^= 1;
  }

  const int mb = m_blk + wm * 64 + (lane >> 4) * 8;
  const int nb = n_blk + wn * 32 + (lane & 15);
#pragma unroll
  for (int i = 0; i < 4; ++i)
#pragma unroll
    for (int j = 0; j < 2; ++j)
#pragma unroll
      for (int r = 0; r < 8; ++r)
        C[(long)(mb + i * 16 + r) * N + nb + j * 16] = acc[i][j][r];
}

// ---------------------------------------------------------------------------
// Low-rank gate path, stage 1: t1[row, j] = sum_d x[row,d] * Wgk1[d,j]
// ---------------------------------------------------------------------------
__global__ __launch_bounds__(256) void gk_lowrank1(const float* __restrict__ x,
                                                   const float* __restrict__ W1,
                                                   float* __restrict__ t1) {
  const int row = blockIdx.x;
  const int j = threadIdx.x >> 4, s = threadIdx.x & 15;
  float p = 0.f;
  for (int d = s; d < DD; d += 16) p += x[(long)row * DD + d] * W1[d * LR + j];
  __shared__ float red[256];
  red[threadIdx.x] = p;
  __syncthreads();
  if (s == 0) {
    float a = 0.f;
#pragma unroll
    for (int u = 0; u < 16; ++u) a += red[j * 16 + u];
    t1[(long)row * LR + j] = a;
  }
}

// stage 2: gk = log_sigmoid(t1 @ Wgk2 + bgk) / GN
__global__ __launch_bounds__(256) void gk_lowrank2(const float* __restrict__ t1,
                                                   const float* __restrict__ W2,
                                                   const float* __restrict__ bgk,
                                                   float* __restrict__ gk) {
  long idx = (long)blockIdx.x * 256 + threadIdx.x;
  if (idx >= (long)ROWS * KD) return;
  int r = (int)(idx >> 9), c = (int)(idx & (KD - 1));
  float z = bgk[c];
#pragma unroll
  for (int u = 0; u < LR; ++u) z += t1[(long)r * LR + u] * W2[u * KD + c];
  float ls = fminf(z, 0.f) - log1pf(__expf(-fabsf(z)));
  gk[idx] = ls * INV_GN;
}

// ---------------------------------------------------------------------------
// GLA chunk-scan kernel. Grid = B*H*(DV/64) = 32 blocks, 256 threads (8 waves).
// State S (128x64 f32) lives in LDS for the whole scan. gk/q/k chunk tiles are
// double-buffered in LDS and prefetched one chunk ahead via the Tensor Data
// Mover (TENSORcnt-tracked) so DMA overlaps the WMMA stages of the previous
// chunk. Dynamic LDS layout (311,808 bytes, fits 320KB WGP LDS):
// ---------------------------------------------------------------------------
#define OFF_S     0u        // f32 [128][64]   32768
#define OFF_SBF   32768u    // bf16 [64][128]  16384
#define OFF_GLAST 49152u    // f32 [128]       512
#define OFF_QG    49664u    // bf16 [64][128]  16384
#define OFF_KG    66048u    // bf16 [64][128]  16384
#define OFF_KDT   82432u    // bf16 [128][64]  16384
#define OFF_VT    98816u    // bf16 [64][64]   8192
#define OFF_ABF   107008u   // bf16 [64][64]   8192
#define OFF_GK    115200u   // f32 [2][64][128] 65536 (stage, cumsum'd in place)
#define OFF_Q     180736u   // f32 [2][64][128] 65536 (stage)
#define OFF_K     246272u   // f32 [2][64][128] 65536 (stage)
#define GLA_LDS_BYTES 311808u
#define STAGE_BYTES (CC * DK * 4)

__global__ __launch_bounds__(256) void gla_chunk_scan(const float* __restrict__ qf,
                                                      const float* __restrict__ kf,
                                                      const float* __restrict__ vf,
                                                      const float* __restrict__ gkf,
                                                      float* __restrict__ oraw) {
  extern __shared__ char smem[];
  float* S     = (float*)(smem + OFF_S);
  bf16*  Sbf   = (bf16*)(smem + OFF_SBF);
  float* glast = (float*)(smem + OFF_GLAST);
  bf16*  qg    = (bf16*)(smem + OFF_QG);
  bf16*  kg    = (bf16*)(smem + OFF_KG);
  bf16*  kdT   = (bf16*)(smem + OFF_KDT);
  bf16*  vT    = (bf16*)(smem + OFF_VT);
  bf16*  Abf   = (bf16*)(smem + OFF_ABF);

  const int bid = blockIdx.x;
  const int vs = bid & 3;            // DV slice of 64
  const int h  = (bid >> 2) & 3;
  const int b  = bid >> 4;
  const int tid = threadIdx.x, lane = tid & 31, wave = tid >> 5;
  const int qk_col = h * DK;
  const int v_col  = h * DV + vs * 64;

#if HAVE_TDM
  auto issue_chunk = [&](int nn, int tobuf) {
    long rb = (long)b * TT + (long)nn * CC;
    tdm_load_2d_f32(OFF_GK + tobuf * STAGE_BYTES, gkf + rb * KD + qk_col, DK, CC, KD);
    tdm_load_2d_f32(OFF_Q  + tobuf * STAGE_BYTES, qf  + rb * KD + qk_col, DK, CC, KD);
    tdm_load_2d_f32(OFF_K  + tobuf * STAGE_BYTES, kf  + rb * KD + qk_col, DK, CC, KD);
  };
#endif

  for (int i = tid; i < DK * 64; i += 256) S[i] = 0.f;

#if HAVE_TDM
  if (wave == 0) issue_chunk(0, 0);   // prefetch chunk 0
#endif
  __syncthreads();

  for (int n = 0; n < NN; ++n) {
    const int buf = n & 1;
    const long rowBase = (long)b * TT + (long)n * CC;
    float* Gc = (float*)(smem + OFF_GK + buf * STAGE_BYTES);
    const float* Qs = (const float*)(smem + OFF_Q + buf * STAGE_BYTES);
    const float* Ks = (const float*)(smem + OFF_K + buf * STAGE_BYTES);

#if HAVE_TDM
    if (wave == 0) {
      if (n + 1 < NN) {
        issue_chunk(n + 1, buf ^ 1);               // prefetch next chunk
        __builtin_amdgcn_s_wait_tensorcnt(3);      // retire current chunk's 3
      } else {
        __builtin_amdgcn_s_wait_tensorcnt(0);      // last chunk: drain all
      }
    }
#else
    // fallback: cooperative global->LDS copy of the current chunk's tiles
    for (int i = tid; i < CC * DK; i += 256) {
      int c = i >> 7, dk = i & 127;
      Gc[i] = gkf[(rowBase + c) * KD + qk_col + dk];
      ((float*)Qs)[i] = qf[(rowBase + c) * KD + qk_col + dk];
      ((float*)Ks)[i] = kf[(rowBase + c) * KD + qk_col + dk];
    }
#endif
    // (1) snapshot state to bf16 (N-major) for o_inter B-fragments
    for (int i = tid; i < 64 * DK; i += 256) {
      int v = i >> 7, dk = i & 127;
      Sbf[v * DK + dk] = (bf16)S[dk * 64 + v];
    }
    __syncthreads();   // stage tiles + Sbf visible to everyone

    // (2) cumulative sum along the chunk axis per dk column (in place)
    for (int dk = tid; dk < DK; dk += 256) {
      float run = 0.f;
      for (int c = 0; c < CC; ++c) { run += Gc[c * DK + dk]; Gc[c * DK + dk] = run; }
      glast[dk] = run;
    }
    __syncthreads();

    // (3) decayed bf16 tiles + v tile
    for (int i = tid; i < CC * DK; i += 256) {
      int c = i >> 7, dk = i & 127;
      float gv = Gc[i];
      float qv = Qs[i] * QSCALE;
      float kv = Ks[i];
      qg[i] = (bf16)(qv * __expf(gv));
      kg[i] = (bf16)(kv * __expf(-gv));
      kdT[dk * 64 + c] = (bf16)(kv * __expf(glast[dk] - gv));
    }
    for (int i = tid; i < CC * 64; i += 256) {
      int c = i >> 6, v = i & 63;
      vT[v * 64 + c] = (bf16)vf[(rowBase + c) * VD + v_col + v];
    }
    __syncthreads();

    // (4) A = tril(qg @ kg^T): 16 tiles of 16x16, 2 per wave
#pragma unroll
    for (int t = 0; t < 2; ++t) {
      int tl = wave * 2 + t;
      int tm = tl >> 2, tn = tl & 3;
      floatx8 acc = {};
#pragma unroll
      for (int kk = 0; kk < DK; kk += 32)
        acc = wmma_bf16(load_a_frag(qg, DK, tm * 16, kk, lane),
                        load_b_frag(kg, DK, tn * 16, kk, lane), acc);
      int mb = tm * 16 + (lane >> 4) * 8, nb = tn * 16 + (lane & 15);
#pragma unroll
      for (int r = 0; r < 8; ++r) {
        int m = mb + r;
        Abf[m * 64 + nb] = (bf16)((m >= nb) ? acc[r] : 0.f);
      }
    }
    __syncthreads();

    // (5) o = qg @ S_prev + A @ v (inter + intra), 2 tiles per wave
#pragma unroll
    for (int t = 0; t < 2; ++t) {
      int tl = wave * 2 + t;
      int tm = tl >> 2, tn = tl & 3;
      floatx8 acc = {};
#pragma unroll
      for (int kk = 0; kk < DK; kk += 32)
        acc = wmma_bf16(load_a_frag(qg, DK, tm * 16, kk, lane),
                        load_b_frag(Sbf, DK, tn * 16, kk, lane), acc);
#pragma unroll
      for (int kk = 0; kk < CC; kk += 32)
        acc = wmma_bf16(load_a_frag(Abf, 64, tm * 16, kk, lane),
                        load_b_frag(vT, 64, tn * 16, kk, lane), acc);
      int mb = tm * 16 + (lane >> 4) * 8, nb = tn * 16 + (lane & 15);
#pragma unroll
      for (int r = 0; r < 8; ++r)
        oraw[(rowBase + mb + r) * VD + v_col + nb] = acc[r];
    }

    // (6) S = exp(glast) (.)_dk S + kd^T @ v : 32 tiles, 4 per wave
    //     per-row decay folded into the WMMA C accumulator load.
#pragma unroll
    for (int t = 0; t < 4; ++t) {
      int tl = wave * 4 + t;
      int tm = tl >> 2, tn = tl & 3;
      int mb = tm * 16 + (lane >> 4) * 8, nb = tn * 16 + (lane & 15);
      floatx8 acc;
#pragma unroll
      for (int r = 0; r < 8; ++r) {
        int dk = mb + r;
        acc[r] = S[dk * 64 + nb] * __expf(glast[dk]);
      }
#pragma unroll
      for (int kk = 0; kk < CC; kk += 32)
        acc = wmma_bf16(load_a_frag(kdT, 64, tm * 16, kk, lane),
                        load_b_frag(vT, 64, tn * 16, kk, lane), acc);
#pragma unroll
      for (int r = 0; r < 8; ++r) S[(mb + r) * 64 + nb] = acc[r];
    }
    __syncthreads();   // next chunk may snapshot S / refill stage buffers
  }
}

// ---------------------------------------------------------------------------
// RMS norm over DV per (b,t,h) + SiLU(g) gate + norm_w, cast to bf16
// ---------------------------------------------------------------------------
__global__ __launch_bounds__(256) void rms_gate(const float* __restrict__ oraw,
                                                const float* __restrict__ gf,
                                                const float* __restrict__ norm_w,
                                                bf16* __restrict__ obf) {
  const int grp = blockIdx.x;            // ROWS*H groups
  const int row = grp >> 2, h = grp & 3;
  const int j = threadIdx.x;             // 0..255 = DV
  const long base = (long)row * VD + h * DV;
  float o = oraw[base + j];
  __shared__ float red[256];
  red[j] = o * o;
  __syncthreads();
  for (int s = 128; s > 0; s >>= 1) {
    if (j < s) red[j] += red[j + s];
    __syncthreads();
  }
  float rms = sqrtf(red[0] * (1.0f / DV) + 1e-5f);
  float gv = gf[base + j];
  float silu = gv / (1.0f + __expf(-gv));
  obf[base + j] = (bf16)(o / rms * norm_w[j] * silu);
}

// ---------------------------------------------------------------------------
// Launcher
// ---------------------------------------------------------------------------
extern "C" void kernel_launch(void* const* d_in, const int* in_sizes, int n_in,
                              void* d_out, int out_size, void* d_ws, size_t ws_size,
                              hipStream_t stream) {
  const float* x      = (const float*)d_in[0];
  const float* Wq     = (const float*)d_in[1];
  const float* Wk     = (const float*)d_in[2];
  const float* Wv     = (const float*)d_in[3];
  const float* Wg     = (const float*)d_in[4];
  const float* Wgk1   = (const float*)d_in[5];
  const float* Wgk2   = (const float*)d_in[6];
  const float* bgk    = (const float*)d_in[7];
  const float* norm_w = (const float*)d_in[8];
  const float* Wo     = (const float*)d_in[9];
  float* out = (float*)d_out;

  // workspace carve (256B-aligned regions)
  char* p = (char*)d_ws;
  auto take = [&](size_t bytes) { char* r = p; p += (bytes + 255) & ~(size_t)255; return r; };
  bf16*  xbf   = (bf16*)take((size_t)ROWS * DD * 2);
  bf16*  Wq_b  = (bf16*)take((size_t)DD * KD * 2);
  bf16*  Wk_b  = (bf16*)take((size_t)DD * KD * 2);
  bf16*  Wv_b  = (bf16*)take((size_t)DD * VD * 2);
  bf16*  Wg_b  = (bf16*)take((size_t)DD * VD * 2);
  bf16*  Wo_b  = (bf16*)take((size_t)VD * DD * 2);
  float* qf    = (float*)take((size_t)ROWS * KD * 4);
  float* kf    = (float*)take((size_t)ROWS * KD * 4);
  float* vf    = (float*)take((size_t)ROWS * VD * 4);
  float* gf    = (float*)take((size_t)ROWS * VD * 4);
  float* gk    = (float*)take((size_t)ROWS * KD * 4);
  float* t1    = (float*)take((size_t)ROWS * LR * 4);
  float* oraw  = (float*)take((size_t)ROWS * VD * 4);
  bf16*  obf   = (bf16*)take((size_t)ROWS * VD * 2);

  // 1) casts to bf16
  cast_f32_bf16<<<2048, 256, 0, stream>>>(x,  xbf,  (long)ROWS * DD);
  cast_f32_bf16<<<512,  256, 0, stream>>>(Wq, Wq_b, (long)DD * KD);
  cast_f32_bf16<<<512,  256, 0, stream>>>(Wk, Wk_b, (long)DD * KD);
  cast_f32_bf16<<<1024, 256, 0, stream>>>(Wv, Wv_b, (long)DD * VD);
  cast_f32_bf16<<<1024, 256, 0, stream>>>(Wg, Wg_b, (long)DD * VD);
  cast_f32_bf16<<<1024, 256, 0, stream>>>(Wo, Wo_b, (long)VD * DD);

  // 2) projection GEMMs (M=8192, K=1024), 128x128 block tiles
  dim3 blk(256);
  gemm_bf16_wmma<<<dim3(KD / 128, ROWS / 128), blk, 0, stream>>>(xbf, Wq_b, qf, ROWS, KD, DD);
  gemm_bf16_wmma<<<dim3(KD / 128, ROWS / 128), blk, 0, stream>>>(xbf, Wk_b, kf, ROWS, KD, DD);
  gemm_bf16_wmma<<<dim3(VD / 128, ROWS / 128), blk, 0, stream>>>(xbf, Wv_b, vf, ROWS, VD, DD);
  gemm_bf16_wmma<<<dim3(VD / 128, ROWS / 128), blk, 0, stream>>>(xbf, Wg_b, gf, ROWS, VD, DD);

  // 3) low-rank gate path
  gk_lowrank1<<<ROWS, 256, 0, stream>>>(x, Wgk1, t1);
  gk_lowrank2<<<(ROWS * KD + 255) / 256, 256, 0, stream>>>(t1, Wgk2, bgk, gk);

  // 4) chunked GLA scan with TDM chunk prefetch
  gla_chunk_scan<<<BB * HH * (DV / 64), 256, GLA_LDS_BYTES, stream>>>(qf, kf, vf, gk, oraw);

  // 5) RMS norm + SiLU gating -> bf16
  rms_gate<<<ROWS * HH, 256, 0, stream>>>(oraw, gf, norm_w, obf);

  // 6) output projection
  gemm_bf16_wmma<<<dim3(DD / 128, ROWS / 128), blk, 0, stream>>>(obf, Wo_b, out, ROWS, DD, VD);

  (void)in_sizes; (void)n_in; (void)out_size; (void)ws_size;
}
